// DGCNN_26362509263528
// MI455X (gfx1250) — matrix operand order repaired and checked by reading, not compiled, not run on hardware
//
#include <hip/hip_runtime.h>

typedef __bf16 bf16_t;
typedef __attribute__((ext_vector_type(16))) __bf16 v16bf;
typedef __attribute__((ext_vector_type(8)))  __bf16 v8bf;
typedef __attribute__((ext_vector_type(8)))  float  v8f;

#define BB 8
#define NN 2048
#define CC 768
#define H1 256
#define H2 768
#define EPSV 1e-5f
#define SLOPE 0.2f
#define M_TOTAL (BB*NN)          // 16384 rows total
#define OUT_O_ELEMS (M_TOTAL*H2) // 12582912 floats, then idx ints

// ---------------- async global->LDS staging (CDNA5), guarded with fallback ----------------
#if __has_builtin(__builtin_amdgcn_global_load_async_to_lds_b128) && \
    __has_builtin(__builtin_amdgcn_s_wait_asynccnt)
#define HAVE_ASYNC_LDS 1
typedef int v4i_async __attribute__((vector_size(4 * sizeof(int))));
typedef v4i_async __attribute__((address_space(1))) gv4i;   // global (AS1) int4
typedef v4i_async __attribute__((address_space(3))) lv4i;   // LDS    (AS3) int4
#else
#define HAVE_ASYNC_LDS 0
#endif

__device__ __forceinline__ void stage16B(const bf16_t* g, bf16_t* l) {
#if HAVE_ASYNC_LDS
  // flat->AS1: same 64-bit VA; flat->AS3: low 32 bits are the LDS byte address (ISA 10.2)
  gv4i* gp = (gv4i*)(unsigned long long)(__SIZE_TYPE__)g;
  lv4i* lp = (lv4i*)(unsigned int)(__SIZE_TYPE__)l;
  __builtin_amdgcn_global_load_async_to_lds_b128(gp, lp, 0, 0);
#else
  *(v8bf*)l = *(const v8bf*)g;
#endif
}
__device__ __forceinline__ void stage_wait() {
#if HAVE_ASYNC_LDS
  __builtin_amdgcn_s_wait_asynccnt(0);
#endif
}

// ---------------- WMMA operand loaders (CDNA5 bf16 16x16x32 layouts) ----------------
// A (16x32): lane m=L%16, half=L/16: K runs [kb+8h, +8) and [kb+16+8h, +8)
__device__ __forceinline__ v16bf load_a16(const bf16_t* row, int kb, int half) {
  union { v16bf v; v8bf h[2]; } u;
  u.h[0] = *(const v8bf*)(row + kb + half * 8);
  u.h[1] = *(const v8bf*)(row + kb + 16 + half * 8);
  return u.v;
}
// B (32x16): lane n=L%16, half=L/16: K runs [kb+8h, +8) and [kb+16+8h, +8)
__device__ __forceinline__ v16bf load_b16(const bf16_t* row, int kb, int half) {
  union { v16bf v; v8bf h[2]; } u;
  const bf16_t* p = row + kb + half * 8;
  u.h[0] = *(const v8bf*)(p);
  u.h[1] = *(const v8bf*)(p + 16);
  return u.v;
}

__device__ __forceinline__ void top3_insert(float v, int i,
    float& v0, int& i0, float& v1, int& i1, float& v2, int& i2) {
  // comparator matches jax.lax.top_k tie-breaking (smaller index wins ties)
  bool b0 = (v > v0) || (v == v0 && i < i0);
  bool b1 = (v > v1) || (v == v1 && i < i1);
  bool b2 = (v > v2) || (v == v2 && i < i2);
  if (b0)      { v2 = v1; i2 = i1; v1 = v0; i1 = i0; v0 = v; i0 = i; }
  else if (b1) { v2 = v1; i2 = i1; v1 = v;  i1 = i; }
  else if (b2) { v2 = v;  i2 = i; }
}

__device__ __forceinline__ float leaky(float t) { return t > 0.f ? t : SLOPE * t; }

// ---------------- prep: xp = x + pos, hi/lo bf16 split, row squared norms ----------------
__global__ void prep_kernel(const float* __restrict__ x, const float* __restrict__ pos,
                            bf16_t* __restrict__ xh, bf16_t* __restrict__ xl,
                            float* __restrict__ sq) {
  __shared__ float red[256];
  size_t base = (size_t)blockIdx.x * CC;
  float s = 0.f;
  for (int c = threadIdx.x; c < CC; c += 256) {
    float v = x[base + c] + pos[base + c];
    bf16_t h = (bf16_t)v;
    xh[base + c] = h;
    xl[base + c] = (bf16_t)(v - (float)h);
    s += v * v;
  }
  red[threadIdx.x] = s;
  __syncthreads();
  for (int off = 128; off > 0; off >>= 1) {
    if (threadIdx.x < off) red[threadIdx.x] += red[threadIdx.x + off];
    __syncthreads();
  }
  if (threadIdx.x == 0) sq[blockIdx.x] = red[0];
}

// ---------------- weight prep: w1a = w1[:, :C], w1c = w1[:, C:] - w1a; hi/lo splits ----------------
__global__ void prep_w1_kernel(const float* __restrict__ w1,
                               bf16_t* __restrict__ ah, bf16_t* __restrict__ al,
                               bf16_t* __restrict__ ch, bf16_t* __restrict__ cl) {
  int i = blockIdx.x * 256 + threadIdx.x;   // < H1*CC
  int o = i / CC, c = i % CC;
  float a = w1[(size_t)o * 2 * CC + c];
  float d = w1[(size_t)o * 2 * CC + CC + c] - a;
  bf16_t h = (bf16_t)a; ah[i] = h; al[i] = (bf16_t)(a - (float)h);
  h = (bf16_t)d;        ch[i] = h; cl[i] = (bf16_t)(d - (float)h);
}

__global__ void prep_w2_kernel(const float* __restrict__ w2,
                               bf16_t* __restrict__ wh, bf16_t* __restrict__ wl) {
  int i = blockIdx.x * 256 + threadIdx.x;   // < H2*H1
  float v = w2[i];
  bf16_t h = (bf16_t)v; wh[i] = h; wl[i] = (bf16_t)(v - (float)h);
}

// ---------------- fused distance GEMM + streaming top-3 (KNN) ----------------
// block = 64 threads (2 waves). Block owns a 16-row strip of one batch; waves split column tiles.
// pdist = 2*inner - sq_i - sq_j; compensated bf16: Ah*Bh + Ah*Bl + Al*Bh.
#define KNN_AP 776   // 768 + 8 pad (bf16) -> conflict-free ds_load_b128
__global__ void __launch_bounds__(64) knn_kernel(const bf16_t* __restrict__ xh,
                                                 const bf16_t* __restrict__ xl,
                                                 const float* __restrict__ sq,
                                                 int* __restrict__ idx_out) {
  extern __shared__ char smem[];
  bf16_t* Ah    = (bf16_t*)smem;                       // 16 x 776 bf16
  bf16_t* Al    = (bf16_t*)(smem + 24832);             // 16 x 776 bf16
  float*  sqrow = (float*)(smem + 49664);              // 16
  float*  dtile = (float*)(smem + 49728);              // 2 waves x 16 x 18 f32
  float*  mval  = (float*)(smem + 52032);              // 2 x 16 x 3
  int*    midx  = (int*)  (smem + 52416);              // 2 x 16 x 3

  const int tid  = threadIdx.x;
  const int wave = tid >> 5;
  const int lane = tid & 31;
  const int m16  = lane & 15;
  const int half = lane >> 4;
  const int batch = blockIdx.y;
  const int rowbase = blockIdx.x * 16;
  const size_t pbase = (size_t)batch * NN;

  // stage A strip (16 rows, h+l) into padded LDS (async on CDNA5)
  const bf16_t* gah = xh + (pbase + rowbase) * CC;
  const bf16_t* gal = xl + (pbase + rowbase) * CC;
  for (int i = tid; i < 16 * (CC / 8); i += 64) {
    int r = i / (CC / 8), c8 = (i % (CC / 8)) * 8;
    stage16B(gah + (size_t)r * CC + c8, Ah + r * KNN_AP + c8);
    stage16B(gal + (size_t)r * CC + c8, Al + r * KNN_AP + c8);
  }
  if (tid < 16) sqrow[tid] = sq[pbase + rowbase + tid];
  stage_wait();
  __syncthreads();

  float tv0 = -3.4e38f, tv1 = -3.4e38f, tv2 = -3.4e38f;
  int   ti0 = 0x7fffffff, ti1 = 0x7fffffff, ti2 = 0x7fffffff;

  const bf16_t* Arow_h = Ah + m16 * KNN_AP;
  const bf16_t* Arow_l = Al + m16 * KNN_AP;
  float* dt = dtile + wave * 288;   // 16 x 18

  for (int ct = wave; ct < NN / 16; ct += 2) {
    const int colbase = ct * 16;
    const bf16_t* Brow_h = xh + (pbase + colbase + m16) * CC;  // column point's feature row
    const bf16_t* Brow_l = xl + (pbase + colbase + m16) * CC;

    // prefetch next column tile's B rows into cache while this tile computes
    if (ct + 2 < NN / 16) {
      const bf16_t* nh = xh + (pbase + (ct + 2) * 16 + m16) * CC;
      const bf16_t* nl = xl + (pbase + (ct + 2) * 16 + m16) * CC;
      __builtin_prefetch(nh + half * (CC / 2), 0, 1);
      __builtin_prefetch(nl + half * (CC / 2), 0, 1);
    }

    v8f acc = {0.f, 0.f, 0.f, 0.f, 0.f, 0.f, 0.f, 0.f};
    for (int p = 0; p < 3; ++p) {                    // compensated passes
      const bf16_t* ar = (p == 2) ? Arow_l : Arow_h;
      const bf16_t* br = (p == 1) ? Brow_l : Brow_h;
      #pragma unroll 4
      for (int kb = 0; kb < CC; kb += 32) {
        v16bf a = load_a16(ar, kb, half);
        v16bf b = load_b16(br, kb, half);
        acc = __builtin_amdgcn_wmma_f32_16x16x32_bf16(false, a, false, b,
                                                      (short)0, acc, false, false);
      }
    }
    float sqc = sq[pbase + colbase + m16];
    #pragma unroll
    for (int r = 0; r < 8; ++r) {
      int m = half * 8 + r;                          // C/D layout: VGPR r -> M = 8*half + r
      dt[m * 18 + m16] = 2.0f * acc[r] - sqrow[m] - sqc;
    }
    __syncthreads();
    // lane L scans row m16, columns half*8 + 0..7 (ascending index -> stable ties)
    #pragma unroll
    for (int j = 0; j < 8; ++j) {
      int c = half * 8 + j;
      top3_insert(dt[m16 * 18 + c], colbase + c, tv0, ti0, tv1, ti1, tv2, ti2);
    }
    __syncthreads();
  }

  // merge halves within wave
  float ov0 = __shfl_xor(tv0, 16); int oi0 = __shfl_xor(ti0, 16);
  float ov1 = __shfl_xor(tv1, 16); int oi1 = __shfl_xor(ti1, 16);
  float ov2 = __shfl_xor(tv2, 16); int oi2 = __shfl_xor(ti2, 16);
  top3_insert(ov0, oi0, tv0, ti0, tv1, ti1, tv2, ti2);
  top3_insert(ov1, oi1, tv0, ti0, tv1, ti1, tv2, ti2);
  top3_insert(ov2, oi2, tv0, ti0, tv1, ti1, tv2, ti2);
  if (lane < 16) {
    int o = (wave * 16 + lane) * 3;
    mval[o] = tv0; mval[o + 1] = tv1; mval[o + 2] = tv2;
    midx[o] = ti0; midx[o + 1] = ti1; midx[o + 2] = ti2;
  }
  __syncthreads();
  // merge across waves, write indices
  if (tid < 16) {
    float a0 = mval[tid * 3], a1 = mval[tid * 3 + 1], a2 = mval[tid * 3 + 2];
    int   b0 = midx[tid * 3], b1 = midx[tid * 3 + 1], b2 = midx[tid * 3 + 2];
    int o = (16 + tid) * 3;
    top3_insert(mval[o],     midx[o],     a0, b0, a1, b1, a2, b2);
    top3_insert(mval[o + 1], midx[o + 1], a0, b0, a1, b1, a2, b2);
    top3_insert(mval[o + 2], midx[o + 2], a0, b0, a1, b1, a2, b2);
    int* op = idx_out + (pbase + rowbase + tid) * 3;
    op[0] = b0; op[1] = b1; op[2] = b2;
  }
}

// ---------------- generic compensated-bf16 GEMM: C[M x NOUT] = A[M x K] * B^T, B stored [NOUT x K] ----------------
// block = 64 threads (2 waves), 16-row tile shared, waves split output-column tiles.
__global__ void __launch_bounds__(64) gemm_split_kernel(
    const bf16_t* __restrict__ Agh, const bf16_t* __restrict__ Agl,
    const bf16_t* __restrict__ B0h, const bf16_t* __restrict__ B0l, float* __restrict__ C0,
    const bf16_t* __restrict__ B1h, const bf16_t* __restrict__ B1l, float* __restrict__ C1,
    int Kdim, int NOUT) {
  extern __shared__ char smem[];
  const int KP = Kdim + 8;
  bf16_t* Ah = (bf16_t*)smem;
  bf16_t* Al = Ah + 16 * KP;
  const bf16_t* Bh = blockIdx.y ? B1h : B0h;
  const bf16_t* Bl = blockIdx.y ? B1l : B0l;
  float* Cout = blockIdx.y ? C1 : C0;

  const int tid = threadIdx.x;
  const int wave = tid >> 5;
  const int lane = tid & 31;
  const int m16 = lane & 15;
  const int half = lane >> 4;
  const int rowbase = blockIdx.x * 16;

  const bf16_t* gah = Agh + (size_t)rowbase * Kdim;
  const bf16_t* gal = Agl + (size_t)rowbase * Kdim;
  const int chunks = Kdim / 8;
  for (int i = tid; i < 16 * chunks; i += 64) {
    int r = i / chunks, c8 = (i % chunks) * 8;
    stage16B(gah + (size_t)r * Kdim + c8, Ah + r * KP + c8);
    stage16B(gal + (size_t)r * Kdim + c8, Al + r * KP + c8);
  }
  stage_wait();
  __syncthreads();

  const bf16_t* Arow_h = Ah + m16 * KP;
  const bf16_t* Arow_l = Al + m16 * KP;

  for (int ct = wave; ct < NOUT / 16; ct += 2) {
    const bf16_t* Brow_h = Bh + (size_t)(ct * 16 + m16) * Kdim;
    const bf16_t* Brow_l = Bl + (size_t)(ct * 16 + m16) * Kdim;
    v8f acc = {0.f, 0.f, 0.f, 0.f, 0.f, 0.f, 0.f, 0.f};
    for (int p = 0; p < 3; ++p) {
      const bf16_t* ar = (p == 2) ? Arow_l : Arow_h;
      const bf16_t* br = (p == 1) ? Brow_l : Brow_h;
      #pragma unroll 4
      for (int kb = 0; kb < Kdim; kb += 32) {
        v16bf a = load_a16(ar, kb, half);
        v16bf b = load_b16(br, kb, half);
        acc = __builtin_amdgcn_wmma_f32_16x16x32_bf16(false, a, false, b,
                                                      (short)0, acc, false, false);
      }
    }
    #pragma unroll
    for (int r = 0; r < 8; ++r) {
      int m = half * 8 + r;
      Cout[(size_t)(rowbase + m) * NOUT + ct * 16 + m16] = acc[r];
    }
  }
}

// ---------------- BN1 stats over h[b,n,k,c] = t1[idx] + t2 (no materialization) ----------------
__global__ void stats1_kernel(const float* __restrict__ t1, const float* __restrict__ t2,
                              const int* __restrict__ idx,
                              float* __restrict__ pA, float* __restrict__ pB) {
  int c = threadIdx.x;            // 256 channels
  int blk = blockIdx.x;           // 256 blocks x 64 rows
  float s = 0.f, s2 = 0.f;
  for (int r = 0; r < 64; ++r) {
    int row = blk * 64 + r;
    int batch = row >> 11;
    float tc = t2[(size_t)row * H1 + c];
    #pragma unroll
    for (int k = 0; k < 3; ++k) {
      int j = idx[row * 3 + k];
      float v = t1[((size_t)(batch << 11) + j) * H1 + c] + tc;
      s += v; s2 += v * v;
    }
  }
  pA[(size_t)blk * H1 + c] = s;
  pB[(size_t)blk * H1 + c] = s2;
}

// ---------------- fold BN into scale/shift per channel ----------------
__global__ void reduce_stats_kernel(const float* __restrict__ pA, const float* __restrict__ pB,
                                    int nblk, float cnt,
                                    const float* __restrict__ gamma, const float* __restrict__ beta,
                                    float* __restrict__ scale, float* __restrict__ shift, int nch) {
  int c = blockIdx.x * blockDim.x + threadIdx.x;
  if (c >= nch) return;
  float s = 0.f, s2 = 0.f;
  for (int b = 0; b < nblk; ++b) { s += pA[(size_t)b * nch + c]; s2 += pB[(size_t)b * nch + c]; }
  float m = s / cnt;
  float var = s2 / cnt - m * m;
  float is = rsqrtf(var + EPSV);
  float sc = gamma[c] * is;
  scale[c] = sc;
  shift[c] = beta[c] - m * sc;
}

// ---------------- BN1 apply + leaky + max over k, split result to bf16 hi/lo ----------------
__global__ void apply1_kernel(const float* __restrict__ t1, const float* __restrict__ t2,
                              const int* __restrict__ idx,
                              const float* __restrict__ scale, const float* __restrict__ shift,
                              bf16_t* __restrict__ hh, bf16_t* __restrict__ hl) {
  int c = threadIdx.x;
  int blk = blockIdx.x;
  float sc = scale[c], sh = shift[c];
  for (int r = 0; r < 64; ++r) {
    int row = blk * 64 + r;
    int batch = row >> 11;
    float tc = t2[(size_t)row * H1 + c];
    float mx = -3.4e38f;
    #pragma unroll
    for (int k = 0; k < 3; ++k) {
      int j = idx[row * 3 + k];
      float v = t1[((size_t)(batch << 11) + j) * H1 + c] + tc;
      float y = leaky(v * sc + sh);
      mx = fmaxf(mx, y);
    }
    bf16_t h = (bf16_t)mx;
    hh[(size_t)row * H1 + c] = h;
    hl[(size_t)row * H1 + c] = (bf16_t)(mx - (float)h);
  }
}

// ---------------- BN2 stats over o[b,n,c] ----------------
__global__ void stats2_kernel(const float* __restrict__ ow,
                              float* __restrict__ pA, float* __restrict__ pB) {
  int c = threadIdx.x;            // 768 channels
  int blk = blockIdx.x;           // 256 blocks x 64 rows
  float s = 0.f, s2 = 0.f;
  for (int r = 0; r < 64; ++r) {
    float v = ow[(size_t)(blk * 64 + r) * H2 + c];
    s += v; s2 += v * v;
  }
  pA[(size_t)blk * H2 + c] = s;
  pB[(size_t)blk * H2 + c] = s2;
}

// ---------------- BN2 apply + leaky -> d_out (vectorized float4) ----------------
__global__ void apply2_kernel(const float* __restrict__ ow,
                              const float* __restrict__ scale, const float* __restrict__ shift,
                              float* __restrict__ out) {
  int i = (blockIdx.x * 256 + threadIdx.x) * 4;
  int c = i % H2;
  float4 v = *(const float4*)(ow + i);
  float4 r;
  r.x = leaky(v.x * scale[c]     + shift[c]);
  r.y = leaky(v.y * scale[c + 1] + shift[c + 1]);
  r.z = leaky(v.z * scale[c + 2] + shift[c + 2]);
  r.w = leaky(v.w * scale[c + 3] + shift[c + 3]);
  *(float4*)(out + i) = r;
}

// ---------------- workspace layout (bytes) ----------------
#define OFF_XH   ((size_t)0)            // 25165824  (also reused for OW after gemms)
#define OFF_XL   ((size_t)25165824)     // 25165824
#define OFF_SQ   ((size_t)50331648)     // 65536
#define OFF_W1AH ((size_t)50397184)     // 393216
#define OFF_W1AL ((size_t)50790400)
#define OFF_W1CH ((size_t)51183616)
#define OFF_W1CL ((size_t)51576832)
#define OFF_W2H  ((size_t)51970048)
#define OFF_W2L  ((size_t)52363264)
#define OFF_T1   ((size_t)52756480)     // 16777216
#define OFF_T2   ((size_t)69533696)     // 16777216
#define OFF_HH   ((size_t)86310912)     // 8388608
#define OFF_HL   ((size_t)94699520)     // 8388608
#define OFF_OW   ((size_t)0)            // 50331648 (aliases XH/XL, dead by then)
#define OFF_P1A  ((size_t)103088128)    // 262144
#define OFF_P1B  ((size_t)103350272)    // 262144
#define OFF_SC1  ((size_t)103612416)    // 1024
#define OFF_SH1  ((size_t)103613440)    // 1024
#define OFF_P2A  ((size_t)103614464)    // 786432
#define OFF_P2B  ((size_t)104400896)    // 786432
#define OFF_SC2  ((size_t)105187328)    // 3072
#define OFF_SH2  ((size_t)105190400)    // 3072

extern "C" void kernel_launch(void* const* d_in, const int* in_sizes, int n_in,
                              void* d_out, int out_size, void* d_ws, size_t ws_size,
                              hipStream_t stream) {
  (void)in_sizes; (void)n_in; (void)out_size; (void)ws_size;
  const float* x   = (const float*)d_in[0];
  const float* pos = (const float*)d_in[1];
  const float* w1  = (const float*)d_in[2];
  const float* g1  = (const float*)d_in[3];
  const float* b1  = (const float*)d_in[4];
  const float* w2  = (const float*)d_in[5];
  const float* g2  = (const float*)d_in[6];
  const float* b2  = (const float*)d_in[7];

  char* ws = (char*)d_ws;
  bf16_t* xh   = (bf16_t*)(ws + OFF_XH);
  bf16_t* xl   = (bf16_t*)(ws + OFF_XL);
  float*  sq   = (float*)(ws + OFF_SQ);
  bf16_t* w1ah = (bf16_t*)(ws + OFF_W1AH);
  bf16_t* w1al = (bf16_t*)(ws + OFF_W1AL);
  bf16_t* w1ch = (bf16_t*)(ws + OFF_W1CH);
  bf16_t* w1cl = (bf16_t*)(ws + OFF_W1CL);
  bf16_t* w2h  = (bf16_t*)(ws + OFF_W2H);
  bf16_t* w2l  = (bf16_t*)(ws + OFF_W2L);
  float*  t1   = (float*)(ws + OFF_T1);
  float*  t2   = (float*)(ws + OFF_T2);
  bf16_t* hh   = (bf16_t*)(ws + OFF_HH);
  bf16_t* hl   = (bf16_t*)(ws + OFF_HL);
  float*  ow   = (float*)(ws + OFF_OW);
  float*  p1a  = (float*)(ws + OFF_P1A);
  float*  p1b  = (float*)(ws + OFF_P1B);
  float*  sc1  = (float*)(ws + OFF_SC1);
  float*  sh1  = (float*)(ws + OFF_SH1);
  float*  p2a  = (float*)(ws + OFF_P2A);
  float*  p2b  = (float*)(ws + OFF_P2B);
  float*  sc2  = (float*)(ws + OFF_SC2);
  float*  sh2  = (float*)(ws + OFF_SH2);

  float* out    = (float*)d_out;
  int*   idxout = (int*)(out + OUT_O_ELEMS);

  // 1. xp = x + pos; hi/lo bf16 split; squared norms
  prep_kernel<<<M_TOTAL, 256, 0, stream>>>(x, pos, xh, xl, sq);
  // 2. weight prep
  prep_w1_kernel<<<(H1 * CC) / 256, 256, 0, stream>>>(w1, w1ah, w1al, w1ch, w1cl);
  prep_w2_kernel<<<(H2 * H1) / 256, 256, 0, stream>>>(w2, w2h, w2l);
  // 3. fused pairwise-distance GEMM + top-3 (writes idx into d_out tail)
  knn_kernel<<<dim3(NN / 16, BB), 64, 52800, stream>>>(xh, xl, sq, idxout);
  // 4. t1 = xp*w1a^T, t2 = xp*w1c^T (one launch, y-dim selects)
  gemm_split_kernel<<<dim3(M_TOTAL / 16, 2), 64, 16 * (CC + 8) * 2 * 2, stream>>>(
      xh, xl, w1ah, w1al, t1, w1ch, w1cl, t2, CC, H1);
  // 5. BN1 stats (h = t1[idx] + t2 recomputed on the fly)
  stats1_kernel<<<256, 256, 0, stream>>>(t1, t2, idxout, p1a, p1b);
  reduce_stats_kernel<<<1, 256, 0, stream>>>(p1a, p1b, 256, (float)(M_TOTAL * 3),
                                             g1, b1, sc1, sh1, H1);
  // 6. BN1 apply + leaky + max over k, split to bf16
  apply1_kernel<<<256, 256, 0, stream>>>(t1, t2, idxout, sc1, sh1, hh, hl);
  // 7. o = hmax * w2^T
  gemm_split_kernel<<<dim3(M_TOTAL / 16, 1), 64, 16 * (H1 + 8) * 2 * 2, stream>>>(
      hh, hl, w2h, w2l, ow, w2h, w2l, ow, H1, H2);
  // 8. BN2 stats + finalize
  stats2_kernel<<<256, H2, 0, stream>>>(ow, p2a, p2b);
  reduce_stats_kernel<<<3, 256, 0, stream>>>(p2a, p2b, 256, (float)M_TOTAL,
                                             g2, b2, sc2, sh2, H2);
  apply2_kernel<<<OUT_O_ELEMS / 1024, 256, 0, stream>>>(ow, sc2, sh2, out);
}
